// MyTransformerBlock_22960895164641
// MI455X (gfx1250) — matrix operand order repaired and checked by reading, not compiled, not run on hardware
//
#include <hip/hip_runtime.h>
#include <hip/hip_bf16.h>
#include <cstdint>

typedef __attribute__((ext_vector_type(16))) _Float16 v16h;
typedef __attribute__((ext_vector_type(8)))  _Float16 v8h;
typedef __attribute__((ext_vector_type(8)))  float    v8f;
typedef __attribute__((ext_vector_type(4)))  float    v4f;
typedef __attribute__((ext_vector_type(4)))  int      v4i;

#define WMMA_F16(a, b, c) \
  __builtin_amdgcn_wmma_f32_16x16x32_f16(false, (a), false, (b), (short)0, (c), false, false)

// ---------------------------------------------------------------------------
// CDNA5 async global->LDS copy (ASYNCcnt path), with portable fallback.
// Probe-verified signature: (int4 addrspace(1)* src, int4 addrspace(3)* dst,
//                            imm int offset, imm int cpol)
// ---------------------------------------------------------------------------
#if defined(__gfx1250__) && __has_builtin(__builtin_amdgcn_global_load_async_to_lds_b128)
#define HAVE_ASYNC_LDS 1
#else
#define HAVE_ASYNC_LDS 0
#endif

#if HAVE_ASYNC_LDS
typedef __attribute__((address_space(1))) v4i GlobV4i;
typedef __attribute__((address_space(3))) v4i LdsV4i;
#endif

__device__ __forceinline__ void async_copy_b128(void* lds, const void* gmem) {
#if HAVE_ASYNC_LDS
  __builtin_amdgcn_global_load_async_to_lds_b128(
      (GlobV4i*)(uintptr_t)gmem,
      (LdsV4i*)(uint32_t)(uintptr_t)lds,
      0, 0);
#else
  *(v8h*)lds = *(const v8h*)gmem;
#endif
}

__device__ __forceinline__ void wait_async0() {
#if HAVE_ASYNC_LDS
#if __has_builtin(__builtin_amdgcn_s_wait_asynccnt)
  __builtin_amdgcn_s_wait_asynccnt(0);
#else
  asm volatile("s_wait_asynccnt 0x0" ::: "memory");
#endif
#endif
}

// ---------------------------------------------------------------------------
// Problem constants
// ---------------------------------------------------------------------------
#define BATCH   2
#define SEQ     2048
#define DMODEL  1024
#define NHEADS  16
#define DHEAD   64
#define DFF     4096
#define MTOK    (BATCH * SEQ)     // 4096 tokens

// GEMM tiling
#define BM 128
#define BN 128
#define BK 32
#define LDT 40                    // padded LDS row stride in halves (80B, 16B-aligned)

// ---------------------------------------------------------------------------
// RMSNorm: one block per token, f32 in -> f16 out
// ---------------------------------------------------------------------------
__global__ __launch_bounds__(256)
void rmsnorm_f16_kernel(const float* __restrict__ x, const float* __restrict__ w,
                        _Float16* __restrict__ out, int D) {
  int row = blockIdx.x;
  int tid = threadIdx.x;
  const float* xr = x + (size_t)row * D;
  float ss = 0.f;
  for (int i = tid; i < D; i += 256) { float v = xr[i]; ss += v * v; }
  __shared__ float red[256];
  red[tid] = ss;
  __syncthreads();
  for (int s = 128; s > 0; s >>= 1) {
    if (tid < s) red[tid] += red[tid + s];
    __syncthreads();
  }
  float rinv = rsqrtf(red[0] / (float)D + 1e-5f);
  _Float16* orow = out + (size_t)row * D;
  for (int i = tid; i < D; i += 256)
    orow[i] = (_Float16)(xr[i] * rinv * w[i]);
}

// ---------------------------------------------------------------------------
// RoPE in-place on f16 Q or K. Pairs (2i, 2i+1) within each head of dh=64.
// ---------------------------------------------------------------------------
__global__ __launch_bounds__(256)
void rope_f16_kernel(_Float16* __restrict__ buf, const int* __restrict__ pos, int total) {
  int t = blockIdx.x * blockDim.x + threadIdx.x;
  if (t >= total) return;
  int d2    = t & (DMODEL / 2 - 1);
  int token = t >> 9;
  int h = d2 >> 5;
  int i = d2 & 31;
  float p   = (float)pos[token];
  float inv = __expf(-((float)(2 * i) / (float)DHEAD) * __logf(10000.f));
  float ang = p * inv;
  float c = __cosf(ang), s = __sinf(ang);
  size_t off = (size_t)token * DMODEL + h * DHEAD + 2 * i;
  float x1 = (float)buf[off], x2 = (float)buf[off + 1];
  buf[off]     = (_Float16)(x1 * c - x2 * s);
  buf[off + 1] = (_Float16)(x1 * s + x2 * c);
}

// ---------------------------------------------------------------------------
// WMMA GEMM: C[M,N] = A[M,K](f16,row-major) x W[N,K](f32,row-major)^T
// MODE 0: f16 out.  MODE 1: f32 out + residual.
// 256 threads (8 waves), 128x128 tile, K-step 32, double-buffered LDS.
// A staged via async global->LDS (ASYNCcnt); W prefetched to VGPRs then
// converted f32->f16 into the ping-pong buffer (overlaps WMMA).
// One barrier per K-step.
// ---------------------------------------------------------------------------
template <int MODE>
__global__ __launch_bounds__(256)
void gemm_wmma_kernel(const _Float16* __restrict__ A, const float* __restrict__ W,
                      const float* __restrict__ R, void* __restrict__ Cout,
                      int M, int N, int K) {
  __shared__ __attribute__((aligned(16))) _Float16 As[2][BM * LDT];
  __shared__ __attribute__((aligned(16))) _Float16 Bs[2][BN * LDT];

  int tid   = threadIdx.x;
  int mBase = blockIdx.y * BM;
  int nBase = blockIdx.x * BN;
  int wv    = tid >> 5, lane = tid & 31;
  int wm    = (wv & 1) * 64;
  int wn    = (wv >> 1) * 32;
  int lrow  = lane & 15, lhalf = lane >> 4;
  int srow  = tid >> 1, scol = (tid & 1) * 16;

  v8f acc[4][2];
#pragma unroll
  for (int t = 0; t < 4; ++t)
#pragma unroll
    for (int u = 0; u < 2; ++u)
#pragma unroll
      for (int e = 0; e < 8; ++e) acc[t][u][e] = 0.f;

  auto stageA = [&](int buf, int kk) {
    const _Float16* ag = A + (size_t)(mBase + srow) * K + kk + scol;
    async_copy_b128(&As[buf][srow * LDT + scol], ag);
    async_copy_b128(&As[buf][srow * LDT + scol + 8], ag + 8);
  };
  float wreg[16];
  auto loadW = [&](int kk) {
    const float* wg = W + (size_t)(nBase + srow) * K + kk + scol;
#pragma unroll
    for (int i = 0; i < 16; i += 4) {
      v4f t = *(const v4f*)(wg + i);
      wreg[i] = t.x; wreg[i + 1] = t.y; wreg[i + 2] = t.z; wreg[i + 3] = t.w;
    }
  };
  auto storeW = [&](int buf) {
#pragma unroll
    for (int i = 0; i < 16; ++i) Bs[buf][srow * LDT + scol + i] = (_Float16)wreg[i];
  };

  // prologue: stage tile 0
  stageA(0, 0);
  loadW(0);
  storeW(0);

  int cur = 0;
  for (int kk = 0; kk < K; kk += BK) {
    wait_async0();
    __syncthreads();   // buffer `cur` published; buffer `cur^1` free
    bool next = (kk + BK) < K;
    if (next) {
      stageA(cur ^ 1, kk + BK);   // async, overlaps with WMMA below
      loadW(kk + BK);             // global loads in flight during WMMA
    }

    v16h af[4], bf[2];
#pragma unroll
    for (int t = 0; t < 4; ++t) {
      const _Float16* ap = &As[cur][(wm + t * 16 + lrow) * LDT + lhalf * 16];
      v8h lo = *(const v8h*)ap, hi = *(const v8h*)(ap + 8);
#pragma unroll
      for (int e = 0; e < 8; ++e) { af[t][e] = lo[e]; af[t][e + 8] = hi[e]; }
    }
#pragma unroll
    for (int u = 0; u < 2; ++u)
      bf[u] = *(const v16h*)&Bs[cur][(wn + u * 16 + lrow) * LDT + lhalf * 16];

#pragma unroll
    for (int t = 0; t < 4; ++t)
#pragma unroll
      for (int u = 0; u < 2; ++u)
        acc[t][u] = WMMA_F16(af[t], bf[u], acc[t][u]);

    if (next) storeW(cur ^ 1);    // f32->f16 convert after WMMA issued
    cur ^= 1;
  }

#pragma unroll
  for (int t = 0; t < 4; ++t)
#pragma unroll
    for (int u = 0; u < 2; ++u)
#pragma unroll
      for (int e = 0; e < 8; ++e) {
        int row = mBase + wm + t * 16 + e + 8 * lhalf;
        int col = nBase + wn + u * 16 + lrow;
        size_t idx = (size_t)row * N + col;
        if constexpr (MODE == 0) {
          ((_Float16*)Cout)[idx] = (_Float16)acc[t][u][e];
        } else {
          ((float*)Cout)[idx] = acc[t][u][e] + R[idx];
        }
      }
}

// ---------------------------------------------------------------------------
// Fused FFN gate GEMM: out_f16 = silu(A x W1^T) * (A x W3^T), one K-loop.
// Same double-buffered structure as gemm_wmma_kernel.
// ---------------------------------------------------------------------------
__global__ __launch_bounds__(256)
void gemm_gate_wmma_kernel(const _Float16* __restrict__ A,
                           const float* __restrict__ W1, const float* __restrict__ W3,
                           _Float16* __restrict__ Cout, int M, int N, int K) {
  __shared__ __attribute__((aligned(16))) _Float16 As[2][BM * LDT];
  __shared__ __attribute__((aligned(16))) _Float16 B1s[2][BN * LDT];
  __shared__ __attribute__((aligned(16))) _Float16 B3s[2][BN * LDT];

  int tid   = threadIdx.x;
  int mBase = blockIdx.y * BM;
  int nBase = blockIdx.x * BN;
  int wv    = tid >> 5, lane = tid & 31;
  int wm    = (wv & 1) * 64;
  int wn    = (wv >> 1) * 32;
  int lrow  = lane & 15, lhalf = lane >> 4;
  int srow  = tid >> 1, scol = (tid & 1) * 16;

  v8f accG[4][2], accU[4][2];
#pragma unroll
  for (int t = 0; t < 4; ++t)
#pragma unroll
    for (int u = 0; u < 2; ++u)
#pragma unroll
      for (int e = 0; e < 8; ++e) { accG[t][u][e] = 0.f; accU[t][u][e] = 0.f; }

  auto stageA = [&](int buf, int kk) {
    const _Float16* ag = A + (size_t)(mBase + srow) * K + kk + scol;
    async_copy_b128(&As[buf][srow * LDT + scol], ag);
    async_copy_b128(&As[buf][srow * LDT + scol + 8], ag + 8);
  };
  float wreg1[16], wreg3[16];
  auto loadW = [&](int kk) {
    const float* w1g = W1 + (size_t)(nBase + srow) * K + kk + scol;
    const float* w3g = W3 + (size_t)(nBase + srow) * K + kk + scol;
#pragma unroll
    for (int i = 0; i < 16; i += 4) {
      v4f t1 = *(const v4f*)(w1g + i);
      v4f t3 = *(const v4f*)(w3g + i);
      wreg1[i] = t1.x; wreg1[i + 1] = t1.y; wreg1[i + 2] = t1.z; wreg1[i + 3] = t1.w;
      wreg3[i] = t3.x; wreg3[i + 1] = t3.y; wreg3[i + 2] = t3.z; wreg3[i + 3] = t3.w;
    }
  };
  auto storeW = [&](int buf) {
#pragma unroll
    for (int i = 0; i < 16; ++i) {
      B1s[buf][srow * LDT + scol + i] = (_Float16)wreg1[i];
      B3s[buf][srow * LDT + scol + i] = (_Float16)wreg3[i];
    }
  };

  stageA(0, 0);
  loadW(0);
  storeW(0);

  int cur = 0;
  for (int kk = 0; kk < K; kk += BK) {
    wait_async0();
    __syncthreads();
    bool next = (kk + BK) < K;
    if (next) { stageA(cur ^ 1, kk + BK); loadW(kk + BK); }

    v16h af[4], b1f[2], b3f[2];
#pragma unroll
    for (int t = 0; t < 4; ++t) {
      const _Float16* ap = &As[cur][(wm + t * 16 + lrow) * LDT + lhalf * 16];
      v8h lo = *(const v8h*)ap, hi = *(const v8h*)(ap + 8);
#pragma unroll
      for (int e = 0; e < 8; ++e) { af[t][e] = lo[e]; af[t][e + 8] = hi[e]; }
    }
#pragma unroll
    for (int u = 0; u < 2; ++u) {
      b1f[u] = *(const v16h*)&B1s[cur][(wn + u * 16 + lrow) * LDT + lhalf * 16];
      b3f[u] = *(const v16h*)&B3s[cur][(wn + u * 16 + lrow) * LDT + lhalf * 16];
    }
#pragma unroll
    for (int t = 0; t < 4; ++t)
#pragma unroll
      for (int u = 0; u < 2; ++u) {
        accG[t][u] = WMMA_F16(af[t], b1f[u], accG[t][u]);
        accU[t][u] = WMMA_F16(af[t], b3f[u], accU[t][u]);
      }

    if (next) storeW(cur ^ 1);
    cur ^= 1;
  }

#pragma unroll
  for (int t = 0; t < 4; ++t)
#pragma unroll
    for (int u = 0; u < 2; ++u)
#pragma unroll
      for (int e = 0; e < 8; ++e) {
        int row = mBase + wm + t * 16 + e + 8 * lhalf;
        int col = nBase + wn + u * 16 + lrow;
        float g  = accG[t][u][e];
        float uu = accU[t][u][e];
        float sil = g / (1.f + __expf(-g));
        Cout[(size_t)row * N + col] = (_Float16)(sil * uu);
      }
}

// ---------------------------------------------------------------------------
// Flash attention (causal, online softmax), f16 Q/K/V (RoPE'd), f16 ctx out.
// Grid: (S/64, H, B). Block: 128 threads = 4 waves; wave w owns 16 q rows.
// KV chunks of 32 staged in LDS (K async, V transposed manually).
// ---------------------------------------------------------------------------
__global__ __launch_bounds__(128)
void flash_attn_kernel(const _Float16* __restrict__ Q, const _Float16* __restrict__ Kc,
                       const _Float16* __restrict__ V, _Float16* __restrict__ O) {
  __shared__ __attribute__((aligned(16))) _Float16 Ks[32 * 72];     // [kv][dh] padded
  __shared__ __attribute__((aligned(16))) _Float16 Vt[64 * 40];     // [dh][kv] padded
  __shared__ __attribute__((aligned(16))) _Float16 Pb[4 * 16 * 40]; // per-wave P

  int tid  = threadIdx.x;
  int w    = tid >> 5, lane = tid & 31;
  int lrow = lane & 15, lhalf = lane >> 4;

  int qBase = blockIdx.x * 64;
  int hOff  = blockIdx.y * DHEAD;
  size_t bOff = (size_t)blockIdx.z * SEQ * DMODEL;
  int qRow0 = qBase + w * 16;

  // Q fragments, pre-scaled by 1/sqrt(dh)
  v16h aq[2];
  {
    const _Float16* qp = Q + bOff + (size_t)(qRow0 + lrow) * DMODEL + hOff;
#pragma unroll
    for (int j = 0; j < 2; ++j) {
      const _Float16* p = qp + j * 32 + lhalf * 16;
      v8h lo = *(const v8h*)p, hi = *(const v8h*)(p + 8);
#pragma unroll
      for (int e = 0; e < 8; ++e) {
        aq[j][e]     = (_Float16)((float)lo[e] * 0.125f);
        aq[j][e + 8] = (_Float16)((float)hi[e] * 0.125f);
      }
    }
  }

  v8f o[4];
#pragma unroll
  for (int t = 0; t < 4; ++t)
#pragma unroll
    for (int e = 0; e < 8; ++e) o[t][e] = 0.f;

  float mstat[8], lstat[8];
#pragma unroll
  for (int r = 0; r < 8; ++r) { mstat[r] = -1e30f; lstat[r] = 0.f; }

  _Float16* Pw = &Pb[w * 16 * 40];
  int ss = tid >> 2, sc = (tid & 3) * 16;
  int kvEnd = qBase + 64;   // uniform across the block

  for (int kv = 0; kv < kvEnd; kv += 32) {
    // stage K chunk [32][64] via async path, V transposed [64][32]
    const _Float16* kg = Kc + bOff + (size_t)(kv + ss) * DMODEL + hOff + sc;
    async_copy_b128(&Ks[ss * 72 + sc], kg);
    async_copy_b128(&Ks[ss * 72 + sc + 8], kg + 8);
    const _Float16* vg = V + bOff + (size_t)(kv + ss) * DMODEL + hOff + sc;
    v8h v0 = *(const v8h*)vg, v1 = *(const v8h*)(vg + 8);
#pragma unroll
    for (int i = 0; i < 8; ++i) {
      Vt[(sc + i) * 40 + ss]     = v0[i];
      Vt[(sc + 8 + i) * 40 + ss] = v1[i];
    }
    wait_async0();
    __syncthreads();

    // S = (Q/8) * K^T, two 16-col tiles
    v8f sc2[2];
#pragma unroll
    for (int n0 = 0; n0 < 2; ++n0) {
#pragma unroll
      for (int e = 0; e < 8; ++e) sc2[n0][e] = 0.f;
#pragma unroll
      for (int j = 0; j < 2; ++j) {
        v16h bf = *(const v16h*)&Ks[(n0 * 16 + lrow) * 72 + j * 32 + lhalf * 16];
        sc2[n0] = WMMA_F16(aq[j], bf, sc2[n0]);
      }
    }
    // causal mask
#pragma unroll
    for (int n0 = 0; n0 < 2; ++n0) {
      int colg = kv + n0 * 16 + lrow;
#pragma unroll
      for (int r = 0; r < 8; ++r) {
        int rowg = qRow0 + r + 8 * lhalf;
        if (colg > rowg) sc2[n0][r] = -1e9f;
      }
    }
    // online softmax (row stats live per half-wave; width-16 reductions)
    float fac[8];
#pragma unroll
    for (int r = 0; r < 8; ++r) {
      float mx = fmaxf(sc2[0][r], sc2[1][r]);
#pragma unroll
      for (int d = 1; d < 16; d <<= 1) mx = fmaxf(mx, __shfl_xor(mx, d, 16));
      float mnew = fmaxf(mstat[r], mx);
      fac[r] = __expf(mstat[r] - mnew);
      float p0 = __expf(sc2[0][r] - mnew);
      float p1 = __expf(sc2[1][r] - mnew);
      sc2[0][r] = p0; sc2[1][r] = p1;
      float ps = p0 + p1;
#pragma unroll
      for (int d = 1; d < 16; d <<= 1) ps += __shfl_xor(ps, d, 16);
      lstat[r] = lstat[r] * fac[r] + ps;
      mstat[r] = mnew;
    }
#pragma unroll
    for (int t = 0; t < 4; ++t)
#pragma unroll
      for (int r = 0; r < 8; ++r) o[t][r] *= fac[r];

    // P: C-layout -> row-major LDS -> reload in A-layout
#pragma unroll
    for (int n0 = 0; n0 < 2; ++n0)
#pragma unroll
      for (int r = 0; r < 8; ++r)
        Pw[(r + 8 * lhalf) * 40 + n0 * 16 + lrow] = (_Float16)sc2[n0][r];
    __syncthreads();

    v16h ap;
    {
      const _Float16* pp = &Pw[lrow * 40 + lhalf * 16];
      v8h lo = *(const v8h*)pp, hi = *(const v8h*)(pp + 8);
#pragma unroll
      for (int e = 0; e < 8; ++e) { ap[e] = lo[e]; ap[e + 8] = hi[e]; }
    }
    // O += P * V
#pragma unroll
    for (int t = 0; t < 4; ++t) {
      v16h bv = *(const v16h*)&Vt[(t * 16 + lrow) * 40 + lhalf * 16];
      o[t] = WMMA_F16(ap, bv, o[t]);
    }
    __syncthreads();
  }

  // normalize & store ctx
#pragma unroll
  for (int r = 0; r < 8; ++r) {
    float inv = 1.f / lstat[r];
    int rowg = qRow0 + r + 8 * lhalf;
#pragma unroll
    for (int t = 0; t < 4; ++t)
      O[bOff + (size_t)rowg * DMODEL + hOff + t * 16 + lrow] =
          (_Float16)(o[t][r] * inv);
  }
}

// ---------------------------------------------------------------------------
// Host launcher
// ---------------------------------------------------------------------------
extern "C" void kernel_launch(void* const* d_in, const int* in_sizes, int n_in,
                              void* d_out, int out_size, void* d_ws, size_t ws_size,
                              hipStream_t stream) {
  const float* x    = (const float*)d_in[0];
  const int*   tpos = (const int*)d_in[1];
  const float* q_w  = (const float*)d_in[2];
  const float* k_w  = (const float*)d_in[3];
  const float* v_w  = (const float*)d_in[4];
  const float* o_w  = (const float*)d_in[5];
  const float* ln1  = (const float*)d_in[6];
  const float* ln2  = (const float*)d_in[7];
  const float* w1   = (const float*)d_in[8];
  const float* w2   = (const float*)d_in[9];
  const float* w3   = (const float*)d_in[10];
  float* out = (float*)d_out;

  char* ws = (char*)d_ws;
  size_t off = 0;
  auto take = [&](size_t bytes) -> char* {
    char* p = ws + off;
    off += (bytes + 255) & ~(size_t)255;
    return p;
  };
  _Float16* xn   = (_Float16*)take((size_t)MTOK * DMODEL * 2);
  _Float16* qb   = (_Float16*)take((size_t)MTOK * DMODEL * 2);
  _Float16* kb   = (_Float16*)take((size_t)MTOK * DMODEL * 2);
  _Float16* vb   = (_Float16*)take((size_t)MTOK * DMODEL * 2);
  _Float16* ctx  = (_Float16*)take((size_t)MTOK * DMODEL * 2);
  float*    x1   = (float*)take((size_t)MTOK * DMODEL * 4);
  _Float16* xn2  = (_Float16*)take((size_t)MTOK * DMODEL * 2);
  _Float16* hbuf = (_Float16*)take((size_t)MTOK * DFF * 2);

  dim3 gProj(DMODEL / BN, MTOK / BM);
  dim3 gFfn1(DFF / BN, MTOK / BM);

  rmsnorm_f16_kernel<<<MTOK, 256, 0, stream>>>(x, ln1, xn, DMODEL);
  gemm_wmma_kernel<0><<<gProj, 256, 0, stream>>>(xn, q_w, nullptr, qb, MTOK, DMODEL, DMODEL);
  gemm_wmma_kernel<0><<<gProj, 256, 0, stream>>>(xn, k_w, nullptr, kb, MTOK, DMODEL, DMODEL);
  gemm_wmma_kernel<0><<<gProj, 256, 0, stream>>>(xn, v_w, nullptr, vb, MTOK, DMODEL, DMODEL);
  int nPairs = MTOK * (DMODEL / 2);
  rope_f16_kernel<<<(nPairs + 255) / 256, 256, 0, stream>>>(qb, tpos, nPairs);
  rope_f16_kernel<<<(nPairs + 255) / 256, 256, 0, stream>>>(kb, tpos, nPairs);
  flash_attn_kernel<<<dim3(SEQ / 64, NHEADS, BATCH), 128, 0, stream>>>(qb, kb, vb, ctx);
  gemm_wmma_kernel<1><<<gProj, 256, 0, stream>>>(ctx, o_w, x, x1, MTOK, DMODEL, DMODEL);
  rmsnorm_f16_kernel<<<MTOK, 256, 0, stream>>>(x1, ln2, xn2, DMODEL);
  gemm_gate_wmma_kernel<<<gFfn1, 256, 0, stream>>>(xn2, w1, w3, hbuf, MTOK, DFF, DMODEL);
  gemm_wmma_kernel<1><<<gProj, 256, 0, stream>>>(hbuf, w2, x1, out, MTOK, DMODEL, DFF);
}
